// MAUForecaster_90580860272805
// MI455X (gfx1250) — matrix operand order repaired and checked by reading, not compiled, not run on hardware
//
#include <hip/hip_runtime.h>
#include <cstdint>

typedef __attribute__((ext_vector_type(16))) __bf16 v16bf;
typedef __attribute__((ext_vector_type(8)))  float  v8f;

// ---------------------------------------------------------------- helpers
__device__ __forceinline__ uint32_t f2bf1(float f) {
  union { float f; uint32_t u; } v; v.f = f;
  uint32_t u = v.u;
  if ((u & 0x7fffffffu) > 0x7f800000u) return ((u >> 16) | 0x0040u) & 0xffffu; // NaN
  return ((u + 0x7fffu + ((u >> 16) & 1u)) >> 16) & 0xffffu;                   // RNE
}
__device__ __forceinline__ uint32_t packbf(float a, float b) {
  return f2bf1(a) | (f2bf1(b) << 16);
}
__device__ __forceinline__ float gelu_f(float v) {
  return 0.5f * v * (1.0f + erff(v * 0.70710678118654752440f));
}

// ---------------------------------------------------------------- weight pack
// OIHW fp32 -> bf16 pairs, layout wp[(k/2)*O + o], k = (ky*KW+kx)*I + ci
__global__ void pack_w_k(const float* __restrict__ w, uint32_t* __restrict__ wp,
                         int O, int I, int KH, int KW) {
  int total = O * I * KH * KW / 2;
  for (int idx = blockIdx.x * blockDim.x + threadIdx.x; idx < total;
       idx += gridDim.x * blockDim.x) {
    int o = idx % O;
    int pI = idx / O;
    int k0 = pI * 2;
    int tap = k0 / I;
    int ci = k0 % I;
    int ky = tap / KW, kx = tap % KW;
    float f0 = w[(((size_t)o * I + ci) * KH + ky) * KW + kx];
    float f1 = w[(((size_t)o * I + ci + 1) * KH + ky) * KW + kx];
    wp[idx] = packbf(f0, f1);
  }
}

// ---------------------------------------------------------------- WMMA conv
// in: NHWC fp32 (64x64 spatial, CIN channels), wp: packed bf16 pairs [K/2][64],
// out: NHWC fp32, Cout=64. block = 128 threads (4 waves); wave w owns Cout tile
// w*16..w*16+15 of a 16-pixel M tile. grid.x = B*H*(W/16) = 1024.
// ACT: 0=none, 1=gelu(exact), 2=sigmoid.
template <int KH, int KW, int CIN, int PAD, int ACT>
__global__ __launch_bounds__(128)
void conv_wmma(const float* __restrict__ in, const uint32_t* __restrict__ wp,
               const float* __restrict__ bias, float* __restrict__ out) {
  constexpr int Cout = 64;
  constexpr int Hh = 64, Ww = 64;
  constexpr int CCH = CIN >> 5;               // 32-wide K chunks per tap
  const int lane = threadIdx.x & 31;
  const int wave = threadIdx.x >> 5;
  const int tile = blockIdx.x;
  const int x0 = (tile & 3) << 4;             // W/16 = 4 tiles per row
  const int rem = tile >> 2;
  const int y0 = rem & 63;
  const int b  = rem >> 6;
  const int hl = lane >> 4;                   // half-wave
  const int mr = lane & 15;                   // A row (pixel) / D column (channel)
  const int och = (wave << 4) + mr;

  v8f acc = {};
  for (int ky = 0; ky < KH; ++ky) {
    const int iy = y0 + ky - PAD;
    const bool vy = (iy >= 0) && (iy < Hh);
    const float* rowp = in + ((size_t)(b * Hh + iy) * Ww) * CIN;
#pragma unroll
    for (int kx = 0; kx < KW; ++kx) {
      const int ix = x0 + mr + kx - PAD;
      const bool valid = vy && (ix >= 0) && (ix < Ww);
      const float* src = rowp + (size_t)ix * CIN;
      const int tap = ky * KW + kx;
#pragma unroll
      for (int cb = 0; cb < CCH; ++cb) {
        const int cbase = cb << 5;
        union { v16bf v; uint32_t u[8]; } ua, ub;
        // A fragment: VGPR r holds K = (r&3)*2 + hl*8 + (r>=4)*16 (+cbase)
#pragma unroll
        for (int r = 0; r < 8; ++r) {
          const int k = ((r & 3) << 1) + (hl << 3) + ((r >> 2) << 4) + cbase;
          float fx = 0.f, fy = 0.f;
          if (valid) { float2 f = *(const float2*)(src + k); fx = f.x; fy = f.y; }
          ua.u[r] = packbf(fx, fy);
        }
        // B fragment: lane column n=mr; VGPR r holds K pair ((r&3)*2 + hl*8 + (r>=4)*16, +1)
        const int pairBase = (tap * CIN + cbase) >> 1;
#pragma unroll
        for (int r = 0; r < 8; ++r)
          ub.u[r] = wp[(size_t)(pairBase + (hl << 2) + (r & 3) + ((r >> 2) << 3)) * Cout + och];
        acc = __builtin_amdgcn_wmma_f32_16x16x32_bf16(false, ua.v, false, ub.v,
                                                      (short)0, acc, false, false);
      }
    }
  }
  const float bv = bias[och];
#pragma unroll
  for (int r = 0; r < 8; ++r) {
    const int m = r + (hl << 3);              // D row = output pixel
    float val = acc[r] + bv;
    if (ACT == 1) val = gelu_f(val);
    else if (ACT == 2) val = 1.0f / (1.0f + expf(-val));
    out[((size_t)(b * Hh + y0) * Ww + (x0 + m)) * Cout + och] = val;
  }
}

template <int KH, int KW, int CIN, int PAD, int ACT>
static inline void launch_conv(const float* in, const uint32_t* wp,
                               const float* bias, float* out, hipStream_t s) {
  conv_wmma<KH, KW, CIN, PAD, ACT><<<1024, 128, 0, s>>>(in, wp, bias, out);
}

// ---------------------------------------------------------------- encoder conv1
// x: NCHW slice (5 in-ch), out: NHWC (64 ch), 3x3 pad1, fused exact GELU
__global__ void conv1_enc(const float* __restrict__ x, const float* __restrict__ w,
                          const float* __restrict__ bias, float* __restrict__ out,
                          int xBatchStride) {
  int idx = blockIdx.x * blockDim.x + threadIdx.x;          // B*H*W*64
  if (idx >= 4 * 64 * 64 * 64) return;
  int co = idx & 63;
  int p = idx >> 6;
  int xq = p & 63; p >>= 6;
  int y = p & 63;  int b = p >> 6;
  float s = bias[co];
  for (int ci = 0; ci < 5; ++ci)
#pragma unroll
    for (int ky = 0; ky < 3; ++ky) {
      int iy = y + ky - 1; if (iy < 0 || iy >= 64) continue;
#pragma unroll
      for (int kx = 0; kx < 3; ++kx) {
        int ix = xq + kx - 1; if (ix < 0 || ix >= 64) continue;
        s += x[(size_t)b * xBatchStride + (ci * 64 + iy) * 64 + ix] *
             w[((co * 5 + ci) * 3 + ky) * 3 + kx];
      }
    }
  out[idx] = gelu_f(s);
}

// ---------------------------------------------------------------- GroupNorm(1,C)
__global__ __launch_bounds__(256)
void gnorm_k(const float* __restrict__ in, float* __restrict__ out,
             const float* __restrict__ g, const float* __restrict__ bt, int Npb) {
  int b = blockIdx.x;
  const float* p = in + (size_t)b * Npb;
  float s = 0.f, s2 = 0.f;
  for (int i = threadIdx.x; i < Npb; i += 256) { float v = p[i]; s += v; s2 += v * v; }
  __shared__ float sh1[256], sh2[256];
  sh1[threadIdx.x] = s; sh2[threadIdx.x] = s2; __syncthreads();
  for (int o = 128; o > 0; o >>= 1) {
    if (threadIdx.x < o) { sh1[threadIdx.x] += sh1[threadIdx.x + o];
                           sh2[threadIdx.x] += sh2[threadIdx.x + o]; }
    __syncthreads();
  }
  float mean = sh1[0] / Npb;
  float var = sh2[0] / Npb - mean * mean;
  float rstd = rsqrtf(var + 1e-5f);
  for (int i = threadIdx.x; i < Npb; i += 256) {
    int c = i & 63;
    out[(size_t)b * Npb + i] = (p[i] - mean) * rstd * g[c] + bt[c];
  }
}

// ---------------------------------------------------------------- attention
__global__ __launch_bounds__(256)
void dot5_k(const float* p0, const float* p1, const float* p2, const float* p3,
            const float* p4, const float* __restrict__ sp,
            float* __restrict__ scores, int Npb) {
  const float* ptrs[5] = {p0, p1, p2, p3, p4};
  int j = blockIdx.y, b = blockIdx.x;
  const float* a = ptrs[j] + (size_t)b * Npb;
  const float* q = sp + (size_t)b * Npb;
  float s = 0.f;
  for (int i = threadIdx.x; i < Npb; i += 256) s += a[i] * q[i];
  __shared__ float sh[256];
  sh[threadIdx.x] = s; __syncthreads();
  for (int o = 128; o > 0; o >>= 1) {
    if (threadIdx.x < o) sh[threadIdx.x] += sh[threadIdx.x + o];
    __syncthreads();
  }
  if (threadIdx.x == 0) scores[b * 5 + j] = sh[0];
}

__global__ void softmax5_k(const float* __restrict__ scores, float* __restrict__ attn) {
  int b = threadIdx.x;
  if (b >= 4) return;
  float m = -1e30f;
  for (int j = 0; j < 5; ++j) m = fmaxf(m, scores[b * 5 + j]);
  float e[5], s = 0.f;
  for (int j = 0; j < 5; ++j) { e[j] = expf(scores[b * 5 + j] - m); s += e[j]; }
  for (int j = 0; j < 5; ++j) attn[b * 5 + j] = e[j] / s;
}

__global__ void tatt_k(const float* p0, const float* p1, const float* p2,
                       const float* p3, const float* p4,
                       const float* __restrict__ attn, float* __restrict__ out, int Npb) {
  int idx = blockIdx.x * blockDim.x + threadIdx.x;
  if (idx >= 4 * Npb) return;
  int b = idx / Npb;
  out[idx] = attn[b * 5 + 0] * p0[idx] + attn[b * 5 + 1] * p1[idx] +
             attn[b * 5 + 2] * p2[idx] + attn[b * 5 + 3] * p3[idx] +
             attn[b * 5 + 4] * p4[idx];
}

// out = u*a + (1-u)*b + gamma*c  (out may alias a/b/c elementwise)
__global__ void gate_k(const float* u, const float* a, const float* bq,
                       const float* c, float gamma, float* out, int n) {
  int i = blockIdx.x * blockDim.x + threadIdx.x;
  if (i >= n) return;
  float uu = u[i];
  out[i] = uu * a[i] + (1.0f - uu) * bq[i] + gamma * c[i];
}

__global__ void concat_k(const float* __restrict__ s, const float* __restrict__ t,
                         float* __restrict__ f, int pixels) {
  int idx = blockIdx.x * blockDim.x + threadIdx.x;
  if (idx >= pixels * 128) return;
  int c = idx & 127, p = idx >> 7;
  f[idx] = (c < 64) ? s[p * 64 + c] : t[p * 64 + (c - 64)];
}

// head conv2 1x1 64->24, in NHWC, out NCHW [4][24][64][64]
__global__ void conv_head2(const float* __restrict__ h, const float* __restrict__ w,
                           const float* __restrict__ bias, float* __restrict__ out) {
  int idx = blockIdx.x * blockDim.x + threadIdx.x;   // B*24*H*W
  if (idx >= 4 * 24 * 64 * 64) return;
  int xq = idx & 63; int p = idx >> 6;
  int y = p & 63; p >>= 6;
  int co = p % 24; int b = p / 24;
  const float* hp = h + ((size_t)(b * 64 + y) * 64 + xq) * 64;
  float s = bias[co];
#pragma unroll 8
  for (int ci = 0; ci < 64; ++ci) s += hp[ci] * w[co * 64 + ci];
  out[idx] = s;
}

// ---------------------------------------------------------------- host driver
extern "C" void kernel_launch(void* const* d_in, const int* in_sizes, int n_in,
                              void* d_out, int out_size, void* d_ws, size_t ws_size,
                              hipStream_t stream) {
  (void)in_sizes; (void)n_in; (void)out_size; (void)ws_size;
  const int B = 4, T = 8, H = 64, Wd = 64, TAU = 5;
  const size_t Npb = (size_t)H * Wd * 64;          // 262144 per-sample elems (NHWC C=64)
  const size_t NT = (size_t)B * Npb;               // 1048576
  const size_t TSB = NT * sizeof(float);           // 4 MiB

  // ---- inputs (setup_inputs dict insertion order) ----
  const float* X = (const float*)d_in[0];
  const float* enc_w1 = (const float*)d_in[2];
  const float* enc_b1 = (const float*)d_in[3];
  const float* enc_w2 = (const float*)d_in[4];
  const float* enc_b2 = (const float*)d_in[5];
  const float* enc_g  = (const float*)d_in[6];
  const float* enc_be = (const float*)d_in[7];
  const float *cw[4][8], *cb[4][8], *nt_g[4], *nt_b[4], *ns_g[4], *ns_b[4];
  for (int l = 0; l < 4; ++l) {
    int base = 8 + l * 20;   // ws,wf,wtu,wsu,wtt,wst,wss,wts (W,b pairs), nt(g,b), ns(g,b)
    for (int j = 0; j < 8; ++j) {
      cw[l][j] = (const float*)d_in[base + 2 * j];
      cb[l][j] = (const float*)d_in[base + 2 * j + 1];
    }
    nt_g[l] = (const float*)d_in[base + 16]; nt_b[l] = (const float*)d_in[base + 17];
    ns_g[l] = (const float*)d_in[base + 18]; ns_b[l] = (const float*)d_in[base + 19];
  }
  const float* hw1 = (const float*)d_in[88];
  const float* hb1 = (const float*)d_in[89];
  const float* hw2 = (const float*)d_in[90];
  const float* hb2 = (const float*)d_in[91];

  // ---- workspace bump allocator ----
  char* pws = (char*)d_ws;
  auto alloc = [&](size_t bytes) -> void* {
    void* r = (void*)pws;
    pws += (bytes + 255) & ~(size_t)255;
    return r;
  };
  uint32_t* wpc[4][8];
  for (int l = 0; l < 4; ++l)
    for (int j = 0; j < 8; ++j) wpc[l][j] = (uint32_t*)alloc(800u * 64 * 4);   // 5x5, Cin64
  uint32_t* wpe2 = (uint32_t*)alloc(288u * 64 * 4);                            // 3x3, Cin64
  uint32_t* wph1 = (uint32_t*)alloc(576u * 64 * 4);                            // 3x3, Cin128
  float* zero = (float*)alloc(TSB);
  float* encS[6]; float* tS[4][6]; float* sS[4][6];
  for (int k = 0; k < 6; ++k) encS[k] = (float*)alloc(TSB);
  for (int l = 0; l < 4; ++l)
    for (int k = 0; k < 6; ++k) { tS[l][k] = (float*)alloc(TSB); sS[l][k] = (float*)alloc(TSB); }
  float* tmpA = (float*)alloc(TSB); float* tmpB = (float*)alloc(TSB);
  float* sprime = (float*)alloc(TSB); float* tatt = (float*)alloc(TSB);
  float* uf = (float*)alloc(TSB); float* tami = (float*)alloc(TSB);
  float* ut = (float*)alloc(TSB); float* us = (float*)alloc(TSB);
  float* c1 = (float*)alloc(TSB); float* c2 = (float*)alloc(TSB);
  float* c3 = (float*)alloc(TSB); float* c4 = (float*)alloc(TSB);
  float* fbuf = (float*)alloc(2 * TSB); float* hbuf = (float*)alloc(TSB);
  float* scores = (float*)alloc(256); float* attn = (float*)alloc(256);

  hipMemsetAsync(zero, 0, TSB, stream);

  // ---- pack weights to bf16 pairs ----
  for (int l = 0; l < 4; ++l)
    for (int j = 0; j < 8; ++j)
      pack_w_k<<<200, 256, 0, stream>>>(cw[l][j], wpc[l][j], 64, 64, 5, 5);
  pack_w_k<<<72, 256, 0, stream>>>(enc_w2, wpe2, 64, 64, 3, 3);
  pack_w_k<<<144, 256, 0, stream>>>(hw1, wph1, 64, 128, 3, 3);

  const int GE = (int)(NT / 256);     // 4096 elementwise blocks

  // ---- time loop ----
  for (int t = 0; t < T; ++t) {
    const float* xt = X + (size_t)t * 5 * H * Wd;     // batch stride T*5*H*W
    conv1_enc<<<GE, 256, 0, stream>>>(xt, enc_w1, enc_b1, tmpA, T * 5 * H * Wd);
    launch_conv<3, 3, 64, 1, 1>(tmpA, wpe2, enc_b2, tmpB, stream);
    gnorm_k<<<4, 256, 0, stream>>>(tmpB, encS[t % 6], enc_g, enc_be, (int)Npb);
    const float* cur = encS[t % 6];

    for (int l = 0; l < 4; ++l) {
      const float* sph[5]; const float* tph[5];
      for (int j = 0; j < TAU; ++j) {
        int e = t - TAU + j;
        sph[j] = (e >= 0) ? ((l == 0) ? encS[e % 6] : sS[l - 1][e % 6]) : zero;
        tph[j] = (e >= 0) ? tS[l][e % 6] : zero;
      }
      // s' = conv(s_cur, ws); attn over history; t_att
      launch_conv<5, 5, 64, 2, 0>(cur, wpc[l][0], cb[l][0], sprime, stream);
      dot5_k<<<dim3(4, 5), 256, 0, stream>>>(sph[0], sph[1], sph[2], sph[3], sph[4],
                                             sprime, scores, (int)Npb);
      softmax5_k<<<1, 32, 0, stream>>>(scores, attn);
      tatt_k<<<GE, 256, 0, stream>>>(tph[0], tph[1], tph[2], tph[3], tph[4],
                                     attn, tatt, (int)Npb);
      const float* tprev = tph[4];
      // u_f, t_ami
      launch_conv<5, 5, 64, 2, 2>(tprev, wpc[l][1], cb[l][1], uf, stream);
      gate_k<<<GE, 256, 0, stream>>>(uf, tprev, tatt, tprev, 0.0f, tami, (int)NT);
      // gates u_t, u_s
      launch_conv<5, 5, 64, 2, 2>(tami, wpc[l][2], cb[l][2], ut, stream);
      launch_conv<5, 5, 64, 2, 2>(cur,  wpc[l][3], cb[l][3], us, stream);
      // t_new = gnorm(u_t*conv(t_ami,wtt) + (1-u_t)*conv(s_cur,wst))
      launch_conv<5, 5, 64, 2, 0>(tami, wpc[l][4], cb[l][4], c1, stream);
      launch_conv<5, 5, 64, 2, 0>(cur,  wpc[l][5], cb[l][5], c2, stream);
      gate_k<<<GE, 256, 0, stream>>>(ut, c1, c2, c1, 0.0f, c1, (int)NT);
      // s_new = gnorm(u_s*conv(s_cur,wss) + (1-u_s)*conv(t_ami,wts) + s_cur)
      launch_conv<5, 5, 64, 2, 0>(cur,  wpc[l][6], cb[l][6], c3, stream);
      launch_conv<5, 5, 64, 2, 0>(tami, wpc[l][7], cb[l][7], c4, stream);
      gate_k<<<GE, 256, 0, stream>>>(us, c3, c4, cur, 1.0f, c3, (int)NT);
      gnorm_k<<<4, 256, 0, stream>>>(c1, tS[l][t % 6], nt_g[l], nt_b[l], (int)Npb);
      gnorm_k<<<4, 256, 0, stream>>>(c3, sS[l][t % 6], ns_g[l], ns_b[l], (int)Npb);
      cur = sS[l][t % 6];
    }
  }

  // ---- head ----
  float* topT = tS[3][(T - 1) % 6];
  float* topS = sS[3][(T - 1) % 6];
  concat_k<<<8192, 256, 0, stream>>>(topS, topT, fbuf, B * H * Wd);
  launch_conv<3, 3, 128, 1, 1>(fbuf, wph1, hb1, hbuf, stream);
  conv_head2<<<1536, 256, 0, stream>>>(hbuf, hw2, hb2, (float*)d_out);
}
